// DeepseekV2MLAAttention_31645319037703
// MI455X (gfx1250) — matrix operand order repaired and checked by reading, not compiled, not run on hardware
//
#include <hip/hip_runtime.h>
#include <cmath>

// ---------------------------------------------------------------------------
// DeepSeek-V2 MLA attention for gfx1250 (MI455X), bf16 WMMA pipeline.
//
// Shapes: S=2048 H=5120 N=32 P=128 R=64 V=128 LQ=1536 LKV=512 QK=192
// All GEMMs via v_wmma_f32_16x16x32_bf16 (f32 accumulate); flash-attention
// with online softmax in fp32; async global->LDS staging where available.
// ---------------------------------------------------------------------------

typedef unsigned short u16;
typedef __attribute__((ext_vector_type(16))) __bf16 v16bf;
typedef __attribute__((ext_vector_type(8)))  __bf16 v8bf;
typedef __attribute__((ext_vector_type(8)))  float  v8f;

#define S_  2048
#define H_  5120
#define NH_ 32
#define LQ_ 1536
#define LKV_ 512
#define QK_ 192
#define NQK_ (NH_*QK_)      // 6144
#define KVD_ (NH_*256)      // 8192  (P+V per head)
#define NV_  (NH_*128)      // 4096

#if __has_builtin(__builtin_amdgcn_global_load_async_to_lds_b128)
#define HAVE_ASYNC_LDS 1
#else
#define HAVE_ASYNC_LDS 0
#endif

#if HAVE_ASYNC_LDS
// Builtin signature (from hipcc diagnostic): param 1 is
//   'int __attribute__((vector_size(16))) __device__ *'   (addrspace(1) src)
// param 2 is the LDS destination, then imm offset / imm cpol.
typedef int async_v4i __attribute__((vector_size(16)));
__device__ __forceinline__ void async_copy16(const u16* g, u16* l) {
  __builtin_amdgcn_global_load_async_to_lds_b128(
      (__attribute__((address_space(1))) async_v4i*)g,
      (__attribute__((address_space(3))) async_v4i*)l, 0, 0);
}
#endif

// ---- round-to-nearest-even f32 -> bf16 -------------------------------------
__device__ __forceinline__ u16 f2bf(float f) {
  union { float f; unsigned u; } c; c.f = f;
  unsigned u = c.u;
  return (u16)((u + 0x7FFFu + ((u >> 16) & 1u)) >> 16);
}

__device__ __forceinline__ v16bf cat16(v8bf lo, v8bf hi) {
  return __builtin_shufflevector(lo, hi, 0,1,2,3,4,5,6,7,8,9,10,11,12,13,14,15);
}

// A fragment (16x32 bf16, M x K).  Per ISA 7.12.2:
//   lanes 0-15  : row M=lane,    halfs 0..7 -> K=0..7,  halfs 8..15 -> K=16..23
//   lanes 16-31 : row M=lane-16, halfs 0..7 -> K=8..15, halfs 8..15 -> K=24..31
__device__ __forceinline__ v16bf load_a(const u16* base, int pitch, int lane) {
  int l = lane & 15, hi = lane >> 4;
  const u16* p = base + (size_t)l * pitch;
  v8bf lo = *(const v8bf*)(p + 8 * hi);
  v8bf hh = *(const v8bf*)(p + 16 + 8 * hi);
  return cat16(lo, hh);
}

// B fragment (32x16 bf16, K x N) from an N-major staging buffer:
//   row n holds 32 contiguous K halfs (Bt[n][k]).
//   lanes 0-15: col N=lane, halfs 0..7 -> K=0..7, 8..15 -> K=16..23 ; lanes 16-31: halfs 0..7 -> K=8..15, 8..15 -> K=24..31
__device__ __forceinline__ v16bf load_bt(const u16* base, int pitch, int lane) {
  int n = lane & 15, hi = lane >> 4;
  const u16* p = base + (size_t)n * pitch;
  v8bf lo = *(const v8bf*)(p + 8 * hi);
  v8bf hh = *(const v8bf*)(p + 16 + 8 * hi);
  return cat16(lo, hh);
}

// ---- elementwise f32 -> bf16 ------------------------------------------------
__global__ void cvt_bf16(const float* __restrict__ in, u16* __restrict__ out, size_t n) {
  size_t i = (size_t)blockIdx.x * blockDim.x + threadIdx.x;
  size_t stride = (size_t)gridDim.x * blockDim.x;
  for (; i < n; i += stride) out[i] = f2bf(in[i]);
}

// ---- RMSNorm over first D columns of a row; writes bf16 --------------------
__global__ __launch_bounds__(256) void rmsnorm_bf16(
    const float* __restrict__ x, int xpitch, int D,
    const float* __restrict__ w, u16* __restrict__ out, int opitch) {
  __shared__ float red[8];
  int row = blockIdx.x;
  const float* xr = x + (size_t)row * xpitch;
  float ss = 0.f;
  for (int i = threadIdx.x; i < D; i += 256) { float v = xr[i]; ss += v * v; }
  #pragma unroll
  for (int off = 16; off; off >>= 1) ss += __shfl_xor(ss, off, 32);
  if ((threadIdx.x & 31) == 0) red[threadIdx.x >> 5] = ss;
  __syncthreads();
  if (threadIdx.x == 0) {
    float s = 0.f;
    #pragma unroll
    for (int i = 0; i < 8; ++i) s += red[i];
    red[0] = rsqrtf(s / (float)D + 1e-6f);
  }
  __syncthreads();
  float scale = red[0];
  for (int i = threadIdx.x; i < D; i += 256)
    out[(size_t)row * opitch + i] = f2bf(xr[i] * scale * w[i]);
}

// ---- GEMM: C[M,N] = A[M,K](bf16) * B[K,N](bf16), C f32 ---------------------
// 256 threads (8 waves), 128x128 block tile, double-buffered LDS, K step 32.
// Waves: 4x2 grid -> each wave 32 rows x 64 cols = 8 WMMA accumulators.
// M and K must be multiples of 128/32 (true for all calls); N is guarded.
__global__ __launch_bounds__(256) void gemm_bf16f32(
    const u16* __restrict__ A, const u16* __restrict__ B, float* __restrict__ C,
    int M, int Nn, int K) {
  __shared__ u16 As[2][128][40];   // M x K slab (rows 16B aligned)
  __shared__ u16 Bs[2][128][40];   // N x K slab (transposed: Bs[n][k])
  int tid = threadIdx.x;
  int lane = tid & 31, w = tid >> 5;
  int wrow = w >> 1;               // 0..3 -> M offset 32*wrow
  int wcol = w & 1;                // 0..1 -> N offset 64*wcol
  int m0 = blockIdx.y * 128, n0 = blockIdx.x * 128;

  v8f acc[2][4];
  #pragma unroll
  for (int i = 0; i < 2; ++i)
    #pragma unroll
    for (int j = 0; j < 4; ++j) acc[i][j] = (v8f){};

  auto fill = [&](int buf, int k0) {
    // A: 128 rows x 32 halfs, 2 x 16B chunks per thread
    #pragma unroll
    for (int i = 0; i < 2; ++i) {
      int chunk = tid + 256 * i;               // 0..511
      int r = chunk >> 2, cb = (chunk & 3) * 8;
      const u16* g = A + (size_t)(m0 + r) * K + k0 + cb;
      *(v8bf*)&As[buf][r][cb] = *(const v8bf*)g;
    }
    // B: 32 k-rows x 128 cols -> Bs[n][k] (transpose), 4 cols per thread
    #pragma unroll
    for (int i = 0; i < 4; ++i) {
      int kr = (tid >> 5) + i * 8;             // 0..31
      int nb = (tid & 31) * 4;                 // 0..124
      int col = n0 + nb;
      const u16* g = B + (size_t)(k0 + kr) * Nn + col;
      if (col + 3 < Nn) {
        u16 b0 = g[0], b1 = g[1], b2 = g[2], b3 = g[3];
        Bs[buf][nb][kr] = b0; Bs[buf][nb + 1][kr] = b1;
        Bs[buf][nb + 2][kr] = b2; Bs[buf][nb + 3][kr] = b3;
      } else {
        #pragma unroll
        for (int jj = 0; jj < 4; ++jj)
          Bs[buf][nb + jj][kr] = (col + jj < Nn) ? g[jj] : (u16)0;
      }
    }
  };

  fill(0, 0);
  int cur = 0;
  for (int k0 = 0; k0 < K; k0 += 32) {
    // prefetch two K-steps ahead (global_prefetch_b8)
    if (k0 + 64 < K) {
      __builtin_prefetch(A + (size_t)(m0 + (tid >> 1)) * K + k0 + 64, 0, 0);
      __builtin_prefetch(B + (size_t)(k0 + 64 + (tid >> 5)) * Nn + n0 + (tid & 31) * 4, 0, 0);
    }
    __syncthreads();
    if (k0 + 32 < K) fill(cur ^ 1, k0 + 32);
    v16bf a0 = load_a(&As[cur][32 * wrow][0],      40, lane);
    v16bf a1 = load_a(&As[cur][32 * wrow + 16][0], 40, lane);
    #pragma unroll
    for (int j = 0; j < 4; ++j) {
      v16bf b = load_bt(&Bs[cur][64 * wcol + 16 * j][0], 40, lane);
      acc[0][j] = __builtin_amdgcn_wmma_f32_16x16x32_bf16(false, a0, false, b, (short)0, acc[0][j], false, false);
      acc[1][j] = __builtin_amdgcn_wmma_f32_16x16x32_bf16(false, a1, false, b, (short)0, acc[1][j], false, false);
    }
    cur ^= 1;
  }

  // C layout: VGPR r -> row r + 8*hi, col = lane&15
  int cn = lane & 15, hi = lane >> 4;
  #pragma unroll
  for (int i = 0; i < 2; ++i)
    #pragma unroll
    for (int j = 0; j < 4; ++j)
      #pragma unroll
      for (int r = 0; r < 8; ++r) {
        int row = m0 + 32 * wrow + 16 * i + r + 8 * hi;
        int col = n0 + 64 * wcol + 16 * j + cn;
        if (col < Nn) C[(size_t)row * Nn + col] = acc[i][j][r];
      }
}

// ---- YaRN inverse frequency (R=64, base=1e4, factor=40, orig_max=4096) -----
__device__ __forceinline__ float yarn_inv_freq(int i) {
  const float R = 64.f, BASE = 10000.f, FACTOR = 40.f;
  float pf = powf(BASE, (2.f * (float)i) / R);
  float extrap = 1.f / pf;
  float interp = 1.f / (FACTOR * pf);
  const float TWO_PI = 6.2831853071795864769f;
  float lowf  = floorf(R * logf(4096.f / (32.f * TWO_PI)) / (2.f * logf(BASE)));
  float highf = ceilf (R * logf(4096.f / ( 1.f * TWO_PI)) / (2.f * logf(BASE)));
  lowf = fmaxf(lowf, 0.f);
  highf = fminf(highf, R - 1.f);
  float ramp = ((float)i - lowf) / fmaxf(highf - lowf, 0.001f);
  ramp = fminf(fmaxf(ramp, 0.f), 1.f);
  float mask = 1.f - ramp;
  return interp * (1.f - mask) + extrap * mask;
}

// ---- pack Q: RoPE(gptj) on last 64 dims, *SCALE, head-major bf16 -----------
__global__ __launch_bounds__(256) void pack_q(
    const float* __restrict__ qf, const int* __restrict__ pos,
    u16* __restrict__ Qh, float scale) {
  int s = blockIdx.x;
  __shared__ float cs[32], sn[32];
  if (threadIdx.x < 32) {
    float f = (float)pos[s] * yarn_inv_freq(threadIdx.x);
    cs[threadIdx.x] = cosf(f);
    sn[threadIdx.x] = sinf(f);
  }
  __syncthreads();
  for (int e = threadIdx.x; e < NH_ * QK_; e += 256) {
    int n = e / QK_, d = e % QK_;
    const float* qr = qf + (size_t)s * NQK_ + (size_t)n * QK_;
    float v;
    if (d < 128) v = qr[d];
    else {
      int i = (d - 128) >> 1;
      float x1 = qr[128 + 2 * i], x2 = qr[128 + 2 * i + 1];
      v = ((d & 1) == 0) ? (x1 * cs[i] - x2 * sn[i]) : (x2 * cs[i] + x1 * sn[i]);
    }
    Qh[((size_t)n * S_ + s) * QK_ + d] = f2bf(v * scale);
  }
}

// ---- pack K (k_nope ++ rope(k_pe) broadcast) and V, head-major bf16 --------
__global__ __launch_bounds__(256) void pack_kv(
    const float* __restrict__ latent, const float* __restrict__ kvf,
    const int* __restrict__ pos, u16* __restrict__ Kh, u16* __restrict__ Vh) {
  int s = blockIdx.x;
  __shared__ float kp[64];
  if (threadIdx.x < 32) {
    int i = threadIdx.x;
    float f = (float)pos[s] * yarn_inv_freq(i);
    float c = cosf(f), sv = sinf(f);
    float x1 = latent[(size_t)s * 576 + 512 + 2 * i];
    float x2 = latent[(size_t)s * 576 + 512 + 2 * i + 1];
    kp[2 * i]     = x1 * c - x2 * sv;
    kp[2 * i + 1] = x2 * c + x1 * sv;
  }
  __syncthreads();
  for (int e = threadIdx.x; e < NH_ * QK_; e += 256) {
    int n = e / QK_, d = e % QK_;
    float v = (d < 128) ? kvf[(size_t)s * KVD_ + (size_t)n * 256 + d] : kp[d - 128];
    Kh[((size_t)n * S_ + s) * QK_ + d] = f2bf(v);
  }
  for (int e = threadIdx.x; e < NH_ * 128; e += 256) {
    int n = e >> 7, c = e & 127;
    Vh[((size_t)n * S_ + s) * 128 + c] =
        f2bf(kvf[(size_t)s * KVD_ + (size_t)n * 256 + 128 + c]);
  }
}

// ---- flash attention: block = (64 query rows, 1 head), 4 waves -------------
__global__ __launch_bounds__(128) void mla_attn(
    const u16* __restrict__ Qh, const u16* __restrict__ Kh,
    const u16* __restrict__ Vh, u16* __restrict__ attn) {
  __shared__ u16 Ks[32][200];       // K tile, row-major (== Bt layout for QK^T)
  __shared__ u16 VT[128][40];       // V tile transposed: VT[c][t]
  __shared__ u16 Pw[4][16][40];     // per-wave P staging (C-layout -> A-layout)
  int tid = threadIdx.x, lane = tid & 31, wid = tid >> 5;
  int head = blockIdx.y;
  int q0 = blockIdx.x * 64;
  int l = lane & 15, hi = lane >> 4;

  // Q fragments for this wave's 16 rows (d = 192 -> 6 A fragments)
  const u16* qbase = Qh + ((size_t)head * S_ + q0 + 16 * wid + l) * QK_;
  v16bf qa[6];
  #pragma unroll
  for (int kb = 0; kb < 6; ++kb) {
    const u16* p = qbase + 32 * kb;
    qa[kb] = cat16(*(const v8bf*)(p + 8 * hi), *(const v8bf*)(p + 16 + 8 * hi));
  }

  float mrow[8], lrow[8];
  v8f o[8];
  #pragma unroll
  for (int r = 0; r < 8; ++r) { mrow[r] = -1e30f; lrow[r] = 0.f; }
  #pragma unroll
  for (int j = 0; j < 8; ++j) o[j] = (v8f){};

  int tend = q0 + 64;               // causal: keys <= max row in this block
  for (int t0 = 0; t0 < tend; t0 += 32) {
    // stage K tile: 32 x 192 halfs (async global->LDS when available)
    #pragma unroll
    for (int i = 0; i < 6; ++i) {
      int chunk = tid + 128 * i;                 // 0..767
      int row = chunk / 24, cb = (chunk % 24) * 8;
      const u16* g = Kh + ((size_t)head * S_ + t0 + row) * QK_ + cb;
#if HAVE_ASYNC_LDS
      async_copy16(g, &Ks[row][cb]);
#else
      *(v8bf*)&Ks[row][cb] = *(const v8bf*)g;
#endif
    }
    // stage V tile transposed: 32 x 128
    #pragma unroll
    for (int i = 0; i < 32; ++i) {
      int e = tid + 128 * i;
      int row = e >> 7, col = e & 127;
      VT[col][row] = Vh[((size_t)head * S_ + t0 + row) * 128 + col];
    }
#if HAVE_ASYNC_LDS
    asm volatile("s_wait_asynccnt 0" ::: "memory");
#endif
    __syncthreads();

    // scores: two 16x16 tiles (keys t0..t0+15, t0+16..t0+31)
    v8f s0 = {}, s1 = {};
    #pragma unroll
    for (int kb = 0; kb < 6; ++kb) {
      v16bf b0 = load_bt(&Ks[0][32 * kb], 200, lane);
      s0 = __builtin_amdgcn_wmma_f32_16x16x32_bf16(false, qa[kb], false, b0, (short)0, s0, false, false);
      v16bf b1 = load_bt(&Ks[16][32 * kb], 200, lane);
      s1 = __builtin_amdgcn_wmma_f32_16x16x32_bf16(false, qa[kb], false, b1, (short)0, s1, false, false);
    }

    // causal mask + online softmax.  C-layout: row = r + 8*hi, col = lane&15.
    int rowg_base = q0 + 16 * wid + 8 * hi;
    int c0g = t0 + l, c1g = t0 + 16 + l;
    #pragma unroll
    for (int r = 0; r < 8; ++r) {
      int rowg = rowg_base + r;
      float v0 = (c0g <= rowg) ? s0[r] : -1e30f;
      float v1 = (c1g <= rowg) ? s1[r] : -1e30f;
      float t = fmaxf(v0, v1);
      #pragma unroll
      for (int off = 8; off; off >>= 1) t = fmaxf(t, __shfl_xor(t, off, 32));
      float mn = fmaxf(mrow[r], t);
      float corr = __expf(mrow[r] - mn);
      float p0 = __expf(v0 - mn);
      float p1 = __expf(v1 - mn);
      float psum = p0 + p1;
      #pragma unroll
      for (int off = 8; off; off >>= 1) psum += __shfl_xor(psum, off, 32);
      lrow[r] = lrow[r] * corr + psum;
      mrow[r] = mn;
      #pragma unroll
      for (int j = 0; j < 8; ++j) o[j][r] *= corr;
      Pw[wid][r + 8 * hi][l]      = f2bf(p0);
      Pw[wid][r + 8 * hi][16 + l] = f2bf(p1);
    }
    asm volatile("s_wait_dscnt 0" ::: "memory");   // P write -> P read (same wave)

    // P(16x32) @ V(32x128)
    v16bf pa = load_a(&Pw[wid][0][0], 40, lane);
    #pragma unroll
    for (int jc = 0; jc < 8; ++jc) {
      v16bf bv = load_bt(&VT[16 * jc][0], 40, lane);
      o[jc] = __builtin_amdgcn_wmma_f32_16x16x32_bf16(false, pa, false, bv, (short)0, o[jc], false, false);
    }
    __syncthreads();
  }

  // epilogue: normalize and write attn (S, N*128) as bf16
  #pragma unroll
  for (int r = 0; r < 8; ++r) {
    float inv = 1.f / lrow[r];
    int rowg = q0 + 16 * wid + 8 * hi + r;
    #pragma unroll
    for (int jc = 0; jc < 8; ++jc)
      attn[(size_t)rowg * NV_ + head * 128 + 16 * jc + l] = f2bf(o[jc][r] * inv);
  }
}

// ---------------------------------------------------------------------------
extern "C" void kernel_launch(void* const* d_in, const int* in_sizes, int n_in,
                              void* d_out, int out_size, void* d_ws, size_t ws_size,
                              hipStream_t stream) {
  (void)in_sizes; (void)n_in; (void)out_size; (void)ws_size;
  const int*   positions = (const int*)  d_in[0];
  const float* hidden    = (const float*)d_in[1];
  const float* w_qa      = (const float*)d_in[2];
  const float* q_a_ln_w  = (const float*)d_in[3];
  const float* w_qb      = (const float*)d_in[4];
  const float* w_kva     = (const float*)d_in[5];
  const float* kv_a_ln_w = (const float*)d_in[6];
  const float* w_kvb     = (const float*)d_in[7];
  const float* w_o       = (const float*)d_in[8];
  float* out = (float*)d_out;

  char* ws = (char*)d_ws;
  size_t off = 0;
  auto alloc = [&](size_t bytes) -> char* {
    char* p = ws + off;
    off += (bytes + 255) & ~(size_t)255;
    return p;
  };

  // Phase-A buffers (live through the pre-attention GEMM pipeline)
  u16*   hsB   = (u16*)  alloc((size_t)S_ * H_ * 2);
  u16*   wqaB  = (u16*)  alloc((size_t)H_ * LQ_ * 2);
  float* qlatF = (float*)alloc((size_t)S_ * LQ_ * 4);
  u16*   qlatB = (u16*)  alloc((size_t)S_ * LQ_ * 2);
  u16*   wqbB  = (u16*)  alloc((size_t)LQ_ * NQK_ * 2);
  float* qF    = (float*)alloc((size_t)S_ * NQK_ * 4);
  u16*   wkvaB = (u16*)  alloc((size_t)H_ * 576 * 2);
  float* latF  = (float*)alloc((size_t)S_ * 576 * 4);
  u16*   kvaB  = (u16*)  alloc((size_t)S_ * LKV_ * 2);
  u16*   wkvbB = (u16*)  alloc((size_t)LKV_ * KVD_ * 2);
  float* kvF   = (float*)alloc((size_t)S_ * KVD_ * 4);
  u16*   woB   = (u16*)  alloc((size_t)NV_ * H_ * 2);

  // Phase-B buffers alias the (dead by then) Phase-A prefix:
  //   [hsB .. qF) is dead once pack kernels run -> holds Qh, Kh, Vh.
  //   qF is dead once attention runs             -> holds attn.
  u16* Qh    = (u16*)((char*)hsB);
  u16* Kh    = Qh + (size_t)NH_ * S_ * QK_;
  u16* Vh    = Kh + (size_t)NH_ * S_ * QK_;          // ends < start of qF
  u16* attnB = (u16*)((char*)qF);

  double msc = 0.1 * std::log(40.0) + 1.0;
  float scale = (float)((msc * msc) / std::sqrt(192.0));

  // 1) bf16 conversions of hidden + weights
  cvt_bf16<<<2048, 256, 0, stream>>>(hidden, hsB,   (size_t)S_ * H_);
  cvt_bf16<<<2048, 256, 0, stream>>>(w_qa,   wqaB,  (size_t)H_ * LQ_);
  cvt_bf16<<<2048, 256, 0, stream>>>(w_qb,   wqbB,  (size_t)LQ_ * NQK_);
  cvt_bf16<<<2048, 256, 0, stream>>>(w_kva,  wkvaB, (size_t)H_ * 576);
  cvt_bf16<<<2048, 256, 0, stream>>>(w_kvb,  wkvbB, (size_t)LKV_ * KVD_);
  cvt_bf16<<<2048, 256, 0, stream>>>(w_o,    woB,   (size_t)NV_ * H_);

  // 2) q latent GEMM + RMSNorm
  gemm_bf16f32<<<dim3(LQ_ / 128, S_ / 128), 256, 0, stream>>>(hsB, wqaB, qlatF, S_, LQ_, H_);
  rmsnorm_bf16<<<S_, 256, 0, stream>>>(qlatF, LQ_, LQ_, q_a_ln_w, qlatB, LQ_);

  // 3) q projection GEMM
  gemm_bf16f32<<<dim3(NQK_ / 128, S_ / 128), 256, 0, stream>>>(qlatB, wqbB, qF, S_, NQK_, LQ_);

  // 4) kv latent GEMM + RMSNorm (first 512 of 576; k_pe stays raw in latF)
  gemm_bf16f32<<<dim3((576 + 127) / 128, S_ / 128), 256, 0, stream>>>(hsB, wkvaB, latF, S_, 576, H_);
  rmsnorm_bf16<<<S_, 256, 0, stream>>>(latF, 576, LKV_, kv_a_ln_w, kvaB, LKV_);

  // 5) kv projection GEMM
  gemm_bf16f32<<<dim3(KVD_ / 128, S_ / 128), 256, 0, stream>>>(kvaB, wkvbB, kvF, S_, KVD_, LKV_);

  // 6) RoPE + pack to head-major bf16 (SCALE folded into Q)
  pack_q <<<S_, 256, 0, stream>>>(qF, positions, Qh, scale);
  pack_kv<<<S_, 256, 0, stream>>>(latF, kvF, positions, Kh, Vh);

  // 7) causal flash attention per head
  mla_attn<<<dim3(S_ / 64, NH_), 128, 0, stream>>>(Qh, Kh, Vh, attnB);

  // 8) output projection
  gemm_bf16f32<<<dim3(H_ / 128, S_ / 128), 256, 0, stream>>>(attnB, woB, out, S_, H_, NV_);
}
